// GnnDGL_40072044872259
// MI455X (gfx1250) — compile-verified, matrix operand order checked by reading, not run on hardware
//
#include <hip/hip_runtime.h>

#define NODES  20000
#define EDGES  320000
#define INDIM  16
#define HID    64
#define HEADS  4
#define HD     256      // HEADS*HID
#define NLAYER 2
#define OUTW   208      // 16 + 3*64

typedef __attribute__((ext_vector_type(16))) _Float16 v16h;
typedef __attribute__((ext_vector_type(8)))  _Float16 v8h;
typedef __attribute__((ext_vector_type(8)))  float    v8f;

union V16H { v16h v; v8h h[2]; };

// A-operand (16x32 f16, M x K) per CDNA5 ISA 7.12.2:
// lane L holds row M=L%16; halves 0..7 <- K = b1+0..7, halves 8..15 <- K = b1+16..23,
// where b1 = 0 (lanes 0-15) or 8 (lanes 16-31). Source row-major f16, leading dim ld.
static __device__ __forceinline__ v16h load_a_tile(const _Float16* base, int ld, int row0,
                                                   int ktile, int lane) {
  const int m  = lane & 15;
  const int b1 = (lane < 16) ? 0 : 8;
  const _Float16* p = base + (size_t)(row0 + m) * ld + ktile * 32 + b1;
  V16H a;
  a.h[0] = *(const v8h*)(p);
  a.h[1] = *(const v8h*)(p + 16);
  return a.v;
}

// B-operand tiles pre-packed contiguously: 16 halves per (ktile,ntile,lane).
static __device__ __forceinline__ v16h load_b_tile(const _Float16* packed, int tileidx) {
  const _Float16* p = packed + (size_t)tileidx * 16;
  V16H b;
  b.h[0] = *(const v8h*)(p);
  b.h[1] = *(const v8h*)(p + 8);
  return b.v;
}

static __device__ __forceinline__ v8f wmma_f16(v16h a, v16h b, v8f c) {
  // (neg_a, A, neg_b, B, c_mod, C, reuse_a, reuse_b)
  return __builtin_amdgcn_wmma_f32_16x16x32_f16(false, a, false, b, (short)0, c, false, false);
}

// order-preserving float -> uint key for atomic max over floats
static __device__ __forceinline__ unsigned fkey(float x) {
  unsigned u = __float_as_uint(x);
  return (u & 0x80000000u) ? ~u : (u | 0x80000000u);
}
static __device__ __forceinline__ float fkey_inv(unsigned u) {
  return __uint_as_float((u & 0x80000000u) ? (u & 0x7fffffffu) : ~u);
}

// ---------------- small utility kernels ----------------

__global__ void zero_kernel(unsigned* __restrict__ p, int n) {
  int t = blockIdx.x * blockDim.x + threadIdx.x;
  if (t < n) p[t] = 0u;
}

__global__ void cvt_f16_kernel(const float* __restrict__ s, _Float16* __restrict__ d, int n) {
  int t = blockIdx.x * blockDim.x + threadIdx.x;
  if (t < n) d[t] = (_Float16)s[t];
}

// Pack row-major f32 W (K x NN) into WMMA-B wave32 register layout (f16).
// out[((t*(NN/16)+j)*32+lane)*16 + i] = W[32t + kmap(lane,i)][16j + lane%16]
__global__ void pack_b_kernel(const float* __restrict__ W, _Float16* __restrict__ out,
                              int K, int NN) {
  int idx = blockIdx.x * blockDim.x + threadIdx.x;
  int total = (K >> 5) * (NN >> 4) * 512;
  if (idx >= total) return;
  int i    = idx & 15;
  int lane = (idx >> 4) & 31;
  int rest = idx >> 9;
  int nt   = NN >> 4;
  int j    = rest % nt;
  int t    = rest / nt;
  int kloc = (i < 8) ? (((lane < 16) ? 0 : 8) + i)
                     : (((lane < 16) ? 16 : 24) + (i - 8));
  int k = t * 32 + kloc;
  int n = j * 16 + (lane & 15);
  out[idx] = (_Float16)W[(size_t)k * NN + n];
}

// features = node_feat @ W_embed + b_embed ; also emit f16 copy and output slices
__global__ void embed_kernel(const float* __restrict__ nf, const float* __restrict__ Wemb,
                             const float* __restrict__ bemb, float* __restrict__ feat,
                             _Float16* __restrict__ hf, float* __restrict__ out) {
  int tid = blockIdx.x * blockDim.x + threadIdx.x;
  if (tid >= NODES * HID) return;
  int r = tid >> 6, c = tid & 63;
  float s = bemb[c];
#pragma unroll
  for (int k = 0; k < INDIM; ++k) s += nf[r * INDIM + k] * Wemb[k * HID + c];
  feat[tid] = s;
  hf[tid]   = (_Float16)s;
  out[(size_t)r * OUTW + INDIM + c] = s;
  if (c < INDIM) out[(size_t)r * OUTW + c] = nf[r * INDIM + c];
}

// ---------------- WMMA GEMM kernels ----------------

// f_ni / f_nj / hn : (NODES x 64) @ (64 x 256), one 16x16 tile per wave.
// B traffic here is tiny (2 tiles/wave), no LDS staging needed.
__global__ void gemm_nodeproj_kernel(const _Float16* __restrict__ hf,
                                     const _Float16* __restrict__ Bni,
                                     const _Float16* __restrict__ Bnj,
                                     const _Float16* __restrict__ Bnode,
                                     float* __restrict__ fni, float* __restrict__ fnj,
                                     float* __restrict__ hn) {
  int wid  = (blockIdx.x * blockDim.x + threadIdx.x) >> 5;
  int lane = threadIdx.x & 31;
  const int rowtiles = NODES / 16;
  int mat = wid / (rowtiles * 16);
  if (mat >= 3) return;
  int rem = wid % (rowtiles * 16);
  int rt = rem >> 4, jt = rem & 15;
  const _Float16* Bp = (mat == 0) ? Bni : ((mat == 1) ? Bnj : Bnode);
  float* D = (mat == 0) ? fni : ((mat == 1) ? fnj : hn);
  v8f acc = {};
#pragma unroll
  for (int t = 0; t < 2; ++t) {
    v16h a = load_a_tile(hf, HID, rt * 16, t, lane);
    v16h b = load_b_tile(Bp, (t * 16 + jt) * 32 + lane);
    acc = wmma_f16(a, b, acc);
  }
  int nl = lane & 15, mb = (lane < 16) ? 0 : 8;
  int n = jt * 16 + nl;
#pragma unroll
  for (int v = 0; v < 8; ++v)
    D[(size_t)(rt * 16 + mb + v) * HD + n] = acc[v];
}

// Fused: f_fij GEMM (16 edges x 256) + gather f_ni[src]/f_nj[dst] + bias + leakyrelu
// + dot with attn -> logits, atomic segment-max over dst. One wave per 16-edge tile.
// Packed W_fij (32KB) + attn/bias staged in LDS once per block (8 waves reuse).
__global__ void edge_attn_kernel(const _Float16* __restrict__ ef16,
                                 const _Float16* __restrict__ Bfij,
                                 const float* __restrict__ fni, const float* __restrict__ fnj,
                                 const int* __restrict__ src, const int* __restrict__ dst,
                                 const float* __restrict__ batt, const float* __restrict__ attn,
                                 float* __restrict__ logits, unsigned* __restrict__ mkey) {
  __shared__ _Float16 sB[16384];   // 32 KB packed B
  __shared__ float sAttn[HD];
  __shared__ float sBatt[HD];
  {
    const uint4* g = (const uint4*)Bfij;
    uint4* sp = (uint4*)sB;
    for (int i = threadIdx.x; i < 2048; i += 256) sp[i] = g[i];
    sAttn[threadIdx.x] = attn[threadIdx.x];
    sBatt[threadIdx.x] = batt[threadIdx.x];
  }
  __syncthreads();

  int wid  = (blockIdx.x * blockDim.x + threadIdx.x) >> 5;
  int lane = threadIdx.x & 31;
  if (wid >= EDGES / 16) return;
  int e0 = wid * 16;
  int nl = lane & 15, mb = (lane < 16) ? 0 : 8;
  v16h a0 = load_a_tile(ef16, HID, e0, 0, lane);
  v16h a1 = load_a_tile(ef16, HID, e0, 1, lane);
  int srcs[8], dsts[8];
#pragma unroll
  for (int v = 0; v < 8; ++v) {
    srcs[v] = src[e0 + mb + v];
    dsts[v] = dst[e0 + mb + v];
  }
#pragma unroll
  for (int h = 0; h < HEADS; ++h) {
    float lg[8] = {0.f, 0.f, 0.f, 0.f, 0.f, 0.f, 0.f, 0.f};
#pragma unroll
    for (int jj = 0; jj < 4; ++jj) {
      int j = h * 4 + jj;
      int n = j * 16 + nl;            // n = h*64 + (jj*16+nl)
      v8f acc = {};
      acc = wmma_f16(a0, load_b_tile(sB, (0 * 16 + j) * 32 + lane), acc);
      acc = wmma_f16(a1, load_b_tile(sB, (1 * 16 + j) * 32 + lane), acc);
      float an = sAttn[n];            // attn flat (h*64+d) == n
      float bb = sBatt[n];
#pragma unroll
      for (int v = 0; v < 8; ++v) {
        float x = acc[v] + fni[(size_t)srcs[v] * HD + n] + fnj[(size_t)dsts[v] * HD + n] + bb;
        x = (x > 0.f) ? x : 0.2f * x; // leaky_relu, slope 0.2
        lg[v] += x * an;
      }
    }
    // reduce over 16 N-lanes within each half-wave (rows 0-7 in lanes 0-15, 8-15 in 16-31)
#pragma unroll
    for (int v = 0; v < 8; ++v) {
      lg[v] += __shfl_xor(lg[v], 8, 16);
      lg[v] += __shfl_xor(lg[v], 4, 16);
      lg[v] += __shfl_xor(lg[v], 2, 16);
      lg[v] += __shfl_xor(lg[v], 1, 16);
    }
    if (nl == 0) {
#pragma unroll
      for (int v = 0; v < 8; ++v) {
        int e = e0 + mb + v;
        logits[(size_t)e * HEADS + h] = lg[v];
        atomicMax(&mkey[(size_t)dsts[v] * HEADS + h], fkey(lg[v]));
      }
    }
  }
}

// One wave per edge: w_h = exp(logit - max); z[dst,h] += w; S[dst,:] += w*hn[src,:].
// Each lane owns 8 columns (two float4 loads + 8 atomics); h = lane>>3.
__global__ void edge_scatter_kernel(const float* __restrict__ logits,
                                    const unsigned* __restrict__ mkey,
                                    const int* __restrict__ src, const int* __restrict__ dst,
                                    const float* __restrict__ hn,
                                    float* __restrict__ S, float* __restrict__ z) {
  int wid  = (blockIdx.x * blockDim.x + threadIdx.x) >> 5;
  int lane = threadIdx.x & 31;
  if (wid >= EDGES) return;
  int e = wid;
  int d = dst[e];                 // wave-uniform address: coalesces to one request
  int s = src[e];
  int h = lane >> 3;
  float m = fkey_inv(mkey[(size_t)d * HEADS + h]);
  float w = __expf(logits[(size_t)e * HEADS + h] - m);
  if ((lane & 7) == 0) atomicAdd(&z[(size_t)d * HEADS + h], w);
  const float4* hp = (const float4*)(hn + (size_t)s * HD + lane * 8);
  float4 h0 = hp[0], h1 = hp[1];
  float* Sp = S + (size_t)d * HD + lane * 8;
  atomicAdd(Sp + 0, w * h0.x);
  atomicAdd(Sp + 1, w * h0.y);
  atomicAdd(Sp + 2, w * h0.z);
  atomicAdd(Sp + 3, w * h0.w);
  atomicAdd(Sp + 4, w * h1.x);
  atomicAdd(Sp + 5, w * h1.y);
  atomicAdd(Sp + 6, w * h1.z);
  atomicAdd(Sp + 7, w * h1.w);
}

// h_agg = S / z (f16 for next GEMM); nodes with no in-edges -> 0
__global__ void normalize_kernel(const float* __restrict__ S, const float* __restrict__ z,
                                 _Float16* __restrict__ hagg) {
  int tid = blockIdx.x * blockDim.x + threadIdx.x;
  if (tid >= NODES * HD) return;
  int r = tid >> 8, c = tid & 255, h = c >> 6;
  float zz = z[(size_t)r * HEADS + h];
  float val = (zz > 0.f) ? (S[tid] / zz) : 0.f;
  hagg[tid] = (_Float16)val;
}

// new = h_agg @ W_mlp + b_mlp + features (residual); write f32 features, f16 copy, out slice.
// Packed W_mlp (32KB) staged in LDS per block.
__global__ void gemm_mlp_kernel(const _Float16* __restrict__ hagg,
                                const _Float16* __restrict__ Bmlp,
                                const float* __restrict__ bmlp, float* __restrict__ feat,
                                _Float16* __restrict__ hf, float* __restrict__ out, int outoff) {
  __shared__ _Float16 sB[16384];
  {
    const uint4* g = (const uint4*)Bmlp;
    uint4* sp = (uint4*)sB;
    for (int i = threadIdx.x; i < 2048; i += 256) sp[i] = g[i];
  }
  __syncthreads();

  int wid  = (blockIdx.x * blockDim.x + threadIdx.x) >> 5;
  int lane = threadIdx.x & 31;
  if (wid >= (NODES / 16) * 4) return;
  int rt = wid >> 2, jt = wid & 3;
  v8f acc = {};
#pragma unroll
  for (int t = 0; t < 8; ++t) {
    v16h a = load_a_tile(hagg, HD, rt * 16, t, lane);
    v16h b = load_b_tile(sB, (t * 4 + jt) * 32 + lane);
    acc = wmma_f16(a, b, acc);
  }
  int nl = lane & 15, mb = (lane < 16) ? 0 : 8;
  int n = jt * 16 + nl;
  float bias = bmlp[n];
#pragma unroll
  for (int v = 0; v < 8; ++v) {
    int r = rt * 16 + mb + v;
    float val = acc[v] + bias + feat[(size_t)r * HID + n];
    feat[(size_t)r * HID + n] = val;
    hf[(size_t)r * HID + n]   = (_Float16)val;
    out[(size_t)r * OUTW + outoff + n] = val;
  }
}

// ---------------- host launch ----------------

extern "C" void kernel_launch(void* const* d_in, const int* in_sizes, int n_in,
                              void* d_out, int out_size, void* d_ws, size_t ws_size,
                              hipStream_t stream) {
  (void)in_sizes; (void)n_in; (void)out_size; (void)ws_size;
  const float* node_feat = (const float*)d_in[0];
  const float* edge_feat = (const float*)d_in[1];
  const int*   src       = (const int*)d_in[2];
  const int*   dst       = (const int*)d_in[3];
  const float* W_embed   = (const float*)d_in[4];
  const float* b_embed   = (const float*)d_in[5];
  const float* W_ni      = (const float*)d_in[6];
  const float* W_nj      = (const float*)d_in[7];
  const float* W_fij     = (const float*)d_in[8];
  const float* b_att     = (const float*)d_in[9];
  const float* attn      = (const float*)d_in[10];
  const float* W_node    = (const float*)d_in[11];
  const float* W_mlp     = (const float*)d_in[12];
  const float* b_mlp     = (const float*)d_in[13];
  float* out = (float*)d_out;

  char* ws = (char*)d_ws;
  size_t off = 0;
  auto take = [&](size_t bytes) -> char* {
    char* p = ws + off;
    off = (off + bytes + 255) & ~(size_t)255;
    return p;
  };
  float*    feat   = (float*)take((size_t)NODES * HID * 4);
  _Float16* hf     = (_Float16*)take((size_t)NODES * HID * 2);
  _Float16* ef16   = (_Float16*)take((size_t)EDGES * HID * 2);
  float*    fni    = (float*)take((size_t)NODES * HD * 4);
  float*    fnj    = (float*)take((size_t)NODES * HD * 4);
  float*    hn     = (float*)take((size_t)NODES * HD * 4);
  float*    logits = (float*)take((size_t)EDGES * HEADS * 4);
  unsigned* mkey   = (unsigned*)take((size_t)NODES * HEADS * 4);
  float*    zbuf   = (float*)take((size_t)NODES * HEADS * 4);
  float*    Sbuf   = (float*)take((size_t)NODES * HD * 4);
  _Float16* hagg   = (_Float16*)take((size_t)NODES * HD * 2);
  _Float16* Bni    = (_Float16*)take(16384 * 2);
  _Float16* Bnj    = (_Float16*)take(16384 * 2);
  _Float16* Bfij   = (_Float16*)take(16384 * 2);
  _Float16* Bnode  = (_Float16*)take(16384 * 2);
  _Float16* Bmlp   = (_Float16*)take(16384 * 2);

  const int B = 256;
  embed_kernel<<<(NODES * HID + B - 1) / B, B, 0, stream>>>(node_feat, W_embed, b_embed,
                                                            feat, hf, out);
  cvt_f16_kernel<<<(EDGES * HID + B - 1) / B, B, 0, stream>>>(edge_feat, ef16, EDGES * HID);

  for (int L = 0; L < NLAYER; ++L) {
    const float* WniL   = W_ni   + (size_t)L * HID * HD;
    const float* WnjL   = W_nj   + (size_t)L * HID * HD;
    const float* WfijL  = W_fij  + (size_t)L * HID * HD;
    const float* battL  = b_att  + (size_t)L * HD;
    const float* attnL  = attn   + (size_t)L * HD;
    const float* WnodeL = W_node + (size_t)L * HID * HD;
    const float* WmlpL  = W_mlp  + (size_t)L * HD * HID;
    const float* bmlpL  = b_mlp  + (size_t)L * HID;

    pack_b_kernel<<<64, B, 0, stream>>>(WniL,   Bni,   HID, HD);
    pack_b_kernel<<<64, B, 0, stream>>>(WnjL,   Bnj,   HID, HD);
    pack_b_kernel<<<64, B, 0, stream>>>(WfijL,  Bfij,  HID, HD);
    pack_b_kernel<<<64, B, 0, stream>>>(WnodeL, Bnode, HID, HD);
    pack_b_kernel<<<64, B, 0, stream>>>(WmlpL,  Bmlp,  HD,  HID);

    // 3 GEMMs: f_ni, f_nj, hn (3 * 1250 * 16 waves)
    gemm_nodeproj_kernel<<<(3 * (NODES / 16) * 16 * 32) / B, B, 0, stream>>>(
        hf, Bni, Bnj, Bnode, fni, fnj, hn);

    zero_kernel<<<(NODES * HEADS + B - 1) / B, B, 0, stream>>>(mkey, NODES * HEADS);
    zero_kernel<<<(NODES * HEADS + B - 1) / B, B, 0, stream>>>((unsigned*)zbuf, NODES * HEADS);
    zero_kernel<<<(NODES * HD + B - 1) / B, B, 0, stream>>>((unsigned*)Sbuf, NODES * HD);

    edge_attn_kernel<<<((EDGES / 16) * 32) / B, B, 0, stream>>>(
        ef16, Bfij, fni, fnj, src, dst, battL, attnL, logits, mkey);

    edge_scatter_kernel<<<((size_t)EDGES * 32) / B, B, 0, stream>>>(logits, mkey, src, dst, hn,
                                                                    Sbuf, zbuf);

    normalize_kernel<<<(NODES * HD) / B, B, 0, stream>>>(Sbuf, zbuf, hagg);

    gemm_mlp_kernel<<<((NODES / 16) * 4 * 32) / B, B, 0, stream>>>(
        hagg, Bmlp, bmlpL, feat, hf, out, INDIM + HID + L * HID);
  }
}
